// VectorQuantizer_76235669504494
// MI455X (gfx1250) — matrix operand order repaired and checked by reading, not compiled, not run on hardware
//
#include <hip/hip_runtime.h>
#include <hip/hip_bf16.h>

typedef __attribute__((ext_vector_type(16))) __bf16        v16bf;
typedef __attribute__((ext_vector_type(8)))  float         v8f;
typedef __attribute__((ext_vector_type(8)))  unsigned int  v8u;

union BfPack { v8u u; v16bf b; };

__device__ __forceinline__ unsigned int f2bf1(float f) {
  unsigned int u = __float_as_uint(f);
  return (u + 0x7FFFu + ((u >> 16) & 1u)) >> 16;   // RNE float -> bf16
}
__device__ __forceinline__ unsigned int packbf(float lo, float hi) {
  return f2bf1(lo) | (f2bf1(hi) << 16);
}

// gfx1250 async copy: global -> LDS (16B per lane), tracked by ASYNCcnt
__device__ __forceinline__ void async_load_b128(unsigned lds_off, const void* gaddr) {
  asm volatile("global_load_async_to_lds_b128 %0, %1, off"
               :: "v"(lds_off), "v"((unsigned long long)(uintptr_t)gaddr)
               : "memory");
}
__device__ __forceinline__ void wait_async0() {
  asm volatile("s_wait_asynccnt 0" ::: "memory");
}

// ---------------------------------------------------------------------------
// Kernel 1: emb (fp32) -> bf16 workspace, h[j] = 0.5*||e_j||^2, zero histogram
// grid = NE blocks, block = 256 (= E_DIM)
// ---------------------------------------------------------------------------
__global__ void __launch_bounds__(256)
vq_prep(const float* __restrict__ emb, unsigned short* __restrict__ emb_bf,
        float* __restrict__ h, unsigned int* __restrict__ hist) {
  const int j = blockIdx.x, t = threadIdx.x;
  const float e = emb[(size_t)j * 256 + t];
  emb_bf[(size_t)j * 256 + t] = (unsigned short)f2bf1(e);
  float s = e * e;
  for (int off = 16; off; off >>= 1) s += __shfl_xor(s, off);
  __shared__ float red[8];
  if ((t & 31) == 0) red[t >> 5] = s;
  __syncthreads();
  if (t == 0) {
    float tot = 0.f;
    for (int i = 0; i < 8; ++i) tot += red[i];
    h[j] = 0.5f * tot;
    hist[j] = 0u;
  }
}

// ---------------------------------------------------------------------------
// Kernel 2: bf16 WMMA GEMM + running argmin, LDS-staged B with async double
// buffering. One wave owns a 16-row tile of z; the block's 8 waves share each
// 16-code B tile staged in LDS via global_load_async_to_lds_b128.
// grid = nRows/128 blocks, block = 256 (8 waves -> 8 M-tiles)
// ---------------------------------------------------------------------------
__global__ void __launch_bounds__(256)
vq_argmin(const float* __restrict__ z, const unsigned short* __restrict__ emb_bf,
          const float* __restrict__ h, int* __restrict__ indices, int nTilesN) {
  __shared__ __align__(64) unsigned short smem[2][16 * 256];   // 2 x 8KB

  const int tid  = threadIdx.x;
  const int lane = tid & 31;
  const int wave = tid >> 5;
  const int tile = blockIdx.x * 8 + wave;
  const int rowBase = tile * 16;
  const int half = lane >> 4;          // 0: K+0..15   1: K+16..31
  const int m    = lane & 15;          // A row within tile / B column

  // ---- Build A fragments (16x256 bf16) in contiguous k order (k = 16*half + s) ----
  const float* zrow = z + (size_t)(rowBase + m) * 256;
  v16bf a[8];
#pragma unroll
  for (int kc = 0; kc < 8; ++kc) {
    const float* p1 = zrow + kc * 32 + half * 16;  // K = kc*32 + half*16 + 0..7
    const float4 f0 = *(const float4*)(p1);
    const float4 f1 = *(const float4*)(p1 + 4);
    const float4 f2 = *(const float4*)(p1 + 8);    // K = kc*32+half*16+8..15
    const float4 f3 = *(const float4*)(p1 + 12);
    BfPack u;
    u.u[0] = packbf(f0.x, f0.y); u.u[1] = packbf(f0.z, f0.w);
    u.u[2] = packbf(f1.x, f1.y); u.u[3] = packbf(f1.z, f1.w);
    u.u[4] = packbf(f2.x, f2.y); u.u[5] = packbf(f2.z, f2.w);
    u.u[6] = packbf(f3.x, f3.y); u.u[7] = packbf(f3.z, f3.w);
    a[kc] = u.b;
  }

  // ---- Cooperative staging addresses: thread t moves 32B of each 8KB tile ----
  const char* gbase = (const char*)emb_bf + (size_t)tid * 32;
  const unsigned lds0 = (unsigned)(uintptr_t)(&smem[0][0]) + (unsigned)tid * 32;
  const unsigned lds1 = (unsigned)(uintptr_t)(&smem[1][0]) + (unsigned)tid * 32;

  // prologue: stage tile 0 into buffer 0
  async_load_b128(lds0,      gbase);
  async_load_b128(lds0 + 16, gbase + 16);

  float bestv[8]; int besti[8];
#pragma unroll
  for (int r = 0; r < 8; ++r) { bestv[r] = 3.4e38f; besti[r] = 0; }

  for (int tn = 0; tn < nTilesN; ++tn) {
    wait_async0();        // my wave's staged transfers for buf[tn&1] complete
    __syncthreads();      // everyone's transfers complete, prior reads retired

    if (tn + 1 < nTilesN) {   // overlap next tile's copy with this tile's math
      const char* g = gbase + (size_t)(tn + 1) * 8192;
      const unsigned l = (tn & 1) ? lds0 : lds1;     // buffer (tn+1)&1
      async_load_b128(l,      g);
      async_load_b128(l + 16, g + 16);
    }

    const int n = tn * 16 + m;                    // this lane's column
    const float hn = h[n];
    // B fragments from LDS: lane = column, half selects K 0..15 / 16..31.
    const unsigned short* sb = &smem[tn & 1][0] + (m * 256 + half * 16);
    BfPack ub[8];
#pragma unroll
    for (int kc = 0; kc < 8; ++kc)
      ub[kc].u = *(const v8u*)(sb + kc * 32);      // 32B, 2x ds_load_b128

    // Data-dependency fences: fragments become asm outputs, so the WMMAs
    // (consumers) cannot hoist above, the ds_loads (producers) cannot sink
    // below, and all group members must live in distinct registers.
    asm volatile("" : "+v"(ub[0].u), "+v"(ub[1].u), "+v"(ub[2].u), "+v"(ub[3].u));

    v8f c = {};
#pragma unroll
    for (int kc = 0; kc < 4; ++kc)
      c = __builtin_amdgcn_wmma_f32_16x16x32_bf16(
              false, a[kc], false, ub[kc].b, (short)0, c, false, false);

    asm volatile("" : "+v"(ub[4].u), "+v"(ub[5].u), "+v"(ub[6].u), "+v"(ub[7].u));

#pragma unroll
    for (int kc = 4; kc < 8; ++kc)
      c = __builtin_amdgcn_wmma_f32_16x16x32_bf16(
              false, a[kc], false, ub[kc].b, (short)0, c, false, false);

    // c[r]: row = rowBase + half*8 + r, col = n  -> score = h_n - z.e_n
#pragma unroll
    for (int r = 0; r < 8; ++r) {
      const float s = hn - c[r];
      if (s < bestv[r]) { bestv[r] = s; besti[r] = n; }
    }
  }

  // reduce (value,index) across the 16 lanes of each half (same rows)
#pragma unroll
  for (int r = 0; r < 8; ++r) {
    float v = bestv[r]; int i = besti[r];
    for (int off = 1; off <= 8; off <<= 1) {
      const float ov = __shfl_xor(v, off);
      const int   oi = __shfl_xor(i, off);
      if (ov < v || (ov == v && oi < i)) { v = ov; i = oi; }
    }
    besti[r] = i;
  }
  if (m == 0) {
#pragma unroll
    for (int r = 0; r < 8; ++r) indices[rowBase + half * 8 + r] = besti[r];
  }
}

// ---------------------------------------------------------------------------
// Kernel 3: gather z_q, one-hot rows, SSE partials, histogram, index output
// grid = nRows blocks, block = 256
// ---------------------------------------------------------------------------
__global__ void __launch_bounds__(256)
vq_gather(const float* __restrict__ z, const float* __restrict__ emb,
          const int* __restrict__ indices, float* __restrict__ out,
          float* __restrict__ partials, unsigned int* __restrict__ hist,
          int nRows, int nE) {
  const int row = blockIdx.x;
  const int t   = threadIdx.x;
  const int idx = indices[row];

  const float zv = z[(size_t)row * 256 + t];
  const float qv = emb[(size_t)idx * 256 + t];
  float* zq = out + 1;
  zq[(size_t)row * 256 + t] = qv;                 // z + sg(z_q - z) == z_q

  float d = qv - zv; d *= d;
  for (int off = 16; off; off >>= 1) d += __shfl_xor(d, off);
  __shared__ float red[8];
  if ((t & 31) == 0) red[t >> 5] = d;
  __syncthreads();

  // one-hot row (nE == 1024 -> 4 floats per thread)
  float* oh = out + 2 + (size_t)nRows * 256;
  const int col = t * 4;
  float4 o;
  o.x = (col + 0 == idx) ? 1.f : 0.f;
  o.y = (col + 1 == idx) ? 1.f : 0.f;
  o.z = (col + 2 == idx) ? 1.f : 0.f;
  o.w = (col + 3 == idx) ? 1.f : 0.f;
  *(float4*)(oh + (size_t)row * nE + col) = o;

  if (t == 0) {
    float tot = 0.f;
    for (int i = 0; i < 8; ++i) tot += red[i];
    partials[row] = tot;
    atomicAdd(&hist[idx], 1u);
    float* oidx = out + 2 + (size_t)nRows * 256 + (size_t)nRows * (size_t)nE;
    oidx[row] = (float)idx;
  }
}

// ---------------------------------------------------------------------------
// Kernel 4: deterministic final reduction -> loss, perplexity
// grid = 1 block, block = 1024
// ---------------------------------------------------------------------------
__global__ void __launch_bounds__(1024)
vq_finalize(const float* __restrict__ partials, const unsigned int* __restrict__ hist,
            float* __restrict__ out, int nRows, int nE) {
  const int t = threadIdx.x;
  double ls = 0.0;
  for (int i = t; i < nRows; i += 1024) ls += (double)partials[i];
  float ent = 0.f;
  if (t < nE) {
    const float p = (float)hist[t] / (float)nRows;
    ent = p * logf(p + 1e-10f);
  }
  __shared__ double sd[1024];
  __shared__ float  se[1024];
  sd[t] = ls; se[t] = ent;
  __syncthreads();
  for (int s = 512; s > 0; s >>= 1) {
    if (t < s) { sd[t] += sd[t + s]; se[t] += se[t + s]; }
    __syncthreads();
  }
  if (t == 0) {
    const double mean = sd[0] / ((double)nRows * 256.0);
    out[0] = (float)(1.25 * mean);                    // (1 + BETA) * mse
    out[1 + (size_t)nRows * 256] = expf(-se[0]);      // perplexity
  }
}

// ---------------------------------------------------------------------------
extern "C" void kernel_launch(void* const* d_in, const int* in_sizes, int n_in,
                              void* d_out, int out_size, void* d_ws, size_t ws_size,
                              hipStream_t stream) {
  const float* z   = (const float*)d_in[0];   // (32,2048,256) fp32
  const float* emb = (const float*)d_in[1];   // (1024,256)   fp32

  const int ED    = 256;
  const int NE    = in_sizes[1] / ED;         // 1024
  const int nRows = in_sizes[0] / ED;         // 65536

  auto align256 = [](size_t x) { return (x + 255) & ~(size_t)255; };
  char* ws = (char*)d_ws;
  size_t off = 0;
  unsigned short* emb_bf = (unsigned short*)(ws + off); off = align256(off + (size_t)NE * ED * 2);
  float*          h      = (float*)(ws + off);          off = align256(off + (size_t)NE * 4);
  int*            idxbuf = (int*)(ws + off);            off = align256(off + (size_t)nRows * 4);
  unsigned int*   hist   = (unsigned int*)(ws + off);   off = align256(off + (size_t)NE * 4);
  float*          parts  = (float*)(ws + off);          off = align256(off + (size_t)nRows * 4);
  (void)ws_size; (void)out_size; (void)n_in;

  float* out = (float*)d_out;

  vq_prep   <<<NE,          256, 0, stream>>>(emb, emb_bf, h, hist);
  vq_argmin <<<nRows / 128, 256, 0, stream>>>(z, emb_bf, h, idxbuf, NE / 16);
  vq_gather <<<nRows,       256, 0, stream>>>(z, emb, idxbuf, out, parts, hist, nRows, NE);
  vq_finalize<<<1,         1024, 0, stream>>>(parts, hist, out, nRows, NE);
}
